// DecoderConv_AttenBlock_16569983828336
// MI455X (gfx1250) — matrix operand order repaired and checked by reading, not compiled, not run on hardware
//
#include <hip/hip_runtime.h>
#include <hip/hip_bf16.h>
#include <cstdint>

// ---------------------------------------------------------------------------
// Types for WMMA fragments (gfx1250, wave32)
// ---------------------------------------------------------------------------
typedef _Float16 fp16_t;
typedef __attribute__((ext_vector_type(16))) _Float16 v16h;
typedef __attribute__((ext_vector_type(8)))  _Float16 v8h;
typedef __attribute__((ext_vector_type(8)))  float    v8f;

__device__ __forceinline__ v8f wmma_f16(v16h a, v16h b, v8f c) {
    return __builtin_amdgcn_wmma_f32_16x16x32_f16(
        /*neg_a=*/false, a, /*neg_b=*/false, b,
        /*c_mod=*/(short)0, c, /*reuse_a=*/false, /*reuse_b=*/false);
}

// two aligned 16B loads -> one 16-half fragment
__device__ __forceinline__ v16h ld16(const fp16_t* p) {
    v8h lo = *(const v8h*)p;
    v8h hi = *(const v8h*)(p + 8);
    return __builtin_shufflevector(lo, hi, 0,1,2,3,4,5,6,7,8,9,10,11,12,13,14,15);
}
__device__ __forceinline__ v16h ld8x2(const fp16_t* p0, const fp16_t* p1) {
    v8h lo = *(const v8h*)p0;
    v8h hi = *(const v8h*)p1;
    return __builtin_shufflevector(lo, hi, 0,1,2,3,4,5,6,7,8,9,10,11,12,13,14,15);
}
// low 8 halfs from memory, upper 8 zero (K padded 16->32)
__device__ __forceinline__ v16h ld8z(const fp16_t* p) {
    v8h lo = *(const v8h*)p;
    v8h z  = {};
    return __builtin_shufflevector(lo, z, 0,1,2,3,4,5,6,7,8,9,10,11,12,13,14,15);
}

__device__ __forceinline__ float gelu_exact(float x) {
    return 0.5f * x * (1.0f + erff(x * 0.70710678118654752440f));
}

// ---------------------------------------------------------------------------
// DPP16 row reductions (no LDS): groups are the 16-lane DPP rows.
// quad_perm xor1/xor2, then row_ror:4 and row_ror:8 broadcast the result to
// every lane in the row (commutative ops only).
// ---------------------------------------------------------------------------
template <int CTRL>
__device__ __forceinline__ float dpp_movf(float x) {
    return __int_as_float(__builtin_amdgcn_update_dpp(
        0, __float_as_int(x), CTRL, 0xF, 0xF, true));
}
__device__ __forceinline__ float red_max16(float x) {
    x = fmaxf(x, dpp_movf<0xB1>(x));    // quad_perm(1,0,3,2)  : xor 1
    x = fmaxf(x, dpp_movf<0x4E>(x));    // quad_perm(2,3,0,1)  : xor 2
    x = fmaxf(x, dpp_movf<0x124>(x));   // row_ror:4
    x = fmaxf(x, dpp_movf<0x128>(x));   // row_ror:8
    return x;
}
__device__ __forceinline__ float red_sum16(float x) {
    x += dpp_movf<0xB1>(x);
    x += dpp_movf<0x4E>(x);
    x += dpp_movf<0x124>(x);
    x += dpp_movf<0x128>(x);
    return x;
}

// Problem constants
#define BATCH 2
#define CDIM 128
#define NHEADS 8
#define DH 16
#define GN 2048           // global tokens per batch (8*16*16)
#define NW 256            // windows per batch
#define W3L 64            // local tokens per window
#define W3G 8             // global tokens per global window
#define TOPK_ 4
#define W3 96             // concatenated tokens per window
#define SHW 16384         // 16*32*32

// ---------------------------------------------------------------------------
// f32 [K,N] -> f16 transposed [N,K] (weights)
// ---------------------------------------------------------------------------
__global__ void k_cvt_t(const float* __restrict__ in, fp16_t* __restrict__ out,
                        int K, int N) {
    int i = blockIdx.x * 256 + threadIdx.x;
    if (i < K * N) {
        int n = i / K, k = i % K;
        out[(size_t)n * K + k] = (fp16_t)in[(size_t)k * N + n];
    }
}

// ---------------------------------------------------------------------------
// WMMA GEMM:  out = epilogue(A[M,K](f16) @ BT[N,K](f16)^T)
// grid = (N/64, M/64), block = 128 (4 waves; wave w owns rows i0+16w, 64 cols)
// K compile-time: fully unrolled, software-pipelined.
// ---------------------------------------------------------------------------
template <int K>
__global__ void k_gemm_wmma(const fp16_t* __restrict__ A, const fp16_t* __restrict__ BT,
                            int M, int N,
                            const float* __restrict__ bias,
                            const float* __restrict__ resid,
                            float* __restrict__ outF, fp16_t* __restrict__ outH,
                            int act) {
    const int lane = threadIdx.x & 31;
    const int wv   = threadIdx.x >> 5;
    const int n  = lane & 15;
    const int hs = lane >> 4;
    const int j0 = blockIdx.x * 64;
    const int i0 = blockIdx.y * 64 + wv * 16;
    (void)M;

    const fp16_t* arow = A + (size_t)(i0 + n) * K;
    const fp16_t* brow0 = BT + (size_t)(j0 + n) * K + hs * 16;
    const fp16_t* brow1 = brow0 + (size_t)16 * K;
    const fp16_t* brow2 = brow0 + (size_t)32 * K;
    const fp16_t* brow3 = brow0 + (size_t)48 * K;

    v8f acc[4] = {{}, {}, {}, {}};

    // prologue: load K-step 0
    v16h a_c = ld16(arow + hs * 16);
    v16h b_c0 = ld16(brow0);
    v16h b_c1 = ld16(brow1);
    v16h b_c2 = ld16(brow2);
    v16h b_c3 = ld16(brow3);

#pragma unroll
    for (int k0 = 0; k0 < K; k0 += 32) {
        v16h a_n = {}, b_n0 = {}, b_n1 = {}, b_n2 = {}, b_n3 = {};
        if (k0 + 32 < K) {              // constant-folded per unrolled iter
            a_n  = ld16(arow + k0 + 32 + hs * 16);
            b_n0 = ld16(brow0 + k0 + 32);
            b_n1 = ld16(brow1 + k0 + 32);
            b_n2 = ld16(brow2 + k0 + 32);
            b_n3 = ld16(brow3 + k0 + 32);
        }
        acc[0] = wmma_f16(a_c, b_c0, acc[0]);
        acc[1] = wmma_f16(a_c, b_c1, acc[1]);
        acc[2] = wmma_f16(a_c, b_c2, acc[2]);
        acc[3] = wmma_f16(a_c, b_c3, acc[3]);
        a_c = a_n; b_c0 = b_n0; b_c1 = b_n1; b_c2 = b_n2; b_c3 = b_n3;
    }

#pragma unroll
    for (int jt = 0; jt < 4; ++jt) {
#pragma unroll
        for (int r = 0; r < 8; ++r) {
            int m   = i0 + r + 8 * hs;
            int col = j0 + jt * 16 + n;
            float v = acc[jt][r];
            if (bias)  v += bias[col];
            if (act)   v = gelu_exact(v);
            if (resid) v += resid[(size_t)m * N + col];
            if (outF)  outF[(size_t)m * N + col] = v;
            if (outH)  outH[(size_t)m * N + col] = (fp16_t)v;
        }
    }
}

// ---------------------------------------------------------------------------
// LayerNorm over C=128, one row per 128-thread block. Output row remap:
// orow = (row/gin)*gout + row%gin
// ---------------------------------------------------------------------------
__global__ void k_ln128(const float* __restrict__ x, const float* __restrict__ g,
                        const float* __restrict__ bta, fp16_t* __restrict__ out,
                        int gin, int gout) {
    const int row = blockIdx.x;
    const int c   = threadIdx.x;
    float v = x[(size_t)row * CDIM + c];
    __shared__ float sb[CDIM];
    sb[c] = v;
    __syncthreads();
    for (int s = 64; s > 0; s >>= 1) {
        if (c < s) sb[c] += sb[c + s];
        __syncthreads();
    }
    float mu = sb[0] * (1.0f / CDIM);
    __syncthreads();
    float d = v - mu;
    sb[c] = d * d;
    __syncthreads();
    for (int s = 64; s > 0; s >>= 1) {
        if (c < s) sb[c] += sb[c + s];
        __syncthreads();
    }
    float rs = rsqrtf(sb[0] * (1.0f / CDIM) + 1e-6f);
    int orow = (row / gin) * gout + (row % gin);
    out[(size_t)orow * CDIM + c] = (fp16_t)(d * rs * g[c] + bta[c]);
}

// ---------------------------------------------------------------------------
// Layout builders / scatters
// ---------------------------------------------------------------------------
__global__ void k_build_xg0(const float* __restrict__ xg_in, float* __restrict__ xg0) {
    int r = blockIdx.x;                 // b*2048 + tok
    int c = threadIdx.x;
    int b = r >> 11, tok = r & 2047;
    int gs = tok >> 8, gh = (tok >> 4) & 15, gw = tok & 15;
    xg0[(size_t)r * CDIM + c] =
        xg_in[(((size_t)(b * CDIM + c) * 8 + gs) * 16 + gh) * 16 + gw];
}

__global__ void k_build_shortcut(const float* __restrict__ x_in, float* __restrict__ sl) {
    int r = blockIdx.x;                 // bw*64 + t
    int c = threadIdx.x;
    int bw = r >> 6, t = r & 63;
    int b = bw >> 8, wi = bw & 255;
    int i1 = wi >> 6, i2 = (wi >> 3) & 7, i3 = wi & 7;
    int ms = t >> 4, mh = (t >> 2) & 3, mw = t & 3;
    int s = i1 * 4 + ms, h = i2 * 4 + mh, w = i3 * 4 + mw;
    sl[(size_t)r * CDIM + c] =
        x_in[(((size_t)(b * CDIM + c) * 16 + s) * 32 + h) * 32 + w];
}

__global__ void k_scatter_l(const float* __restrict__ lf, float* __restrict__ outp) {
    int r = blockIdx.x;
    int c = threadIdx.x;
    int bw = r >> 6, t = r & 63;
    int b = bw >> 8, wi = bw & 255;
    int i1 = wi >> 6, i2 = (wi >> 3) & 7, i3 = wi & 7;
    int ms = t >> 4, mh = (t >> 2) & 3, mw = t & 3;
    int s = i1 * 4 + ms, h = i2 * 4 + mh, w = i3 * 4 + mw;
    outp[(((size_t)(b * CDIM + c) * 16 + s) * 32 + h) * 32 + w] =
        lf[(size_t)r * CDIM + c];
}

__global__ void k_scatter_g(const float* __restrict__ xg4, float* __restrict__ outp) {
    int r = blockIdx.x;
    int c = threadIdx.x;
    int b = r >> 11, tok = r & 2047;
    int gs = tok >> 8, gh = (tok >> 4) & 15, gw = tok & 15;
    outp[(((size_t)(b * CDIM + c) * 8 + gs) * 16 + gh) * 16 + gw] =
        xg4[(size_t)r * CDIM + c];
}

// split qkv [4096,384] f16 -> q/k [B,H,2048,16], v transposed [B,H,16,2048]
__global__ void k_split_qkv(const fp16_t* __restrict__ qkvg,
                            fp16_t* __restrict__ q, fp16_t* __restrict__ k,
                            fp16_t* __restrict__ vT) {
    int r = blockIdx.x;                 // token 0..4095
    int j = threadIdx.x;                // 0..383
    int b = r >> 11, nn = r & 2047;
    fp16_t val = qkvg[(size_t)r * 384 + j];
    int which = j >> 7, h = (j >> 4) & 7, d = j & 15;
    if (which == 0)
        q[(((size_t)(b * NHEADS + h)) * GN + nn) * DH + d] = val;
    else if (which == 1)
        k[(((size_t)(b * NHEADS + h)) * GN + nn) * DH + d] = val;
    else
        vT[(((size_t)(b * NHEADS + h)) * DH + d) * GN + nn] = val;
}

// repack local V third of qkvp into [bw, h, d, 96] transposed layout
__global__ void k_repack_vloc(const fp16_t* __restrict__ qkvp, fp16_t* __restrict__ vloc) {
    int bw = blockIdx.x;                // 0..511
    int j  = threadIdx.x;               // 0..127 : h*16+d
    const fp16_t* src = qkvp + (size_t)bw * W3 * 384 + 2 * CDIM + j;
    fp16_t* dst = vloc + ((size_t)bw * CDIM + j) * W3;
    for (int t = 0; t < W3; ++t) dst[t] = src[(size_t)t * 384];
}

// window mean of xg4 -> gwin16 [B*256,128] f16
__global__ void k_gwin_mean(const float* __restrict__ xg4, fp16_t* __restrict__ gwin) {
    int bw = blockIdx.x;
    int c  = threadIdx.x;
    int b = bw >> 8, wi = bw & 255;
    int i1 = wi >> 6, i2 = (wi >> 3) & 7, i3 = wi & 7;
    float s = 0.0f;
#pragma unroll
    for (int tt = 0; tt < 8; ++tt) {
        int ms = tt >> 2, mh = (tt >> 1) & 1, mw = tt & 1;
        int tok = ((i1 * 2 + ms) * 16 + (i2 * 2 + mh)) * 16 + (i3 * 2 + mw);
        s += xg4[((size_t)(b * GN + tok)) * CDIM + c];
    }
    gwin[(size_t)bw * CDIM + c] = (fp16_t)(s * 0.125f);
}

// routing logits + top-4 (scale = DIM^-0.5 applied to qh)
__global__ void k_route_topk(const float* __restrict__ qh, const float* __restrict__ kh,
                             int* __restrict__ tidx) {
    int bw = blockIdx.x;                // b*256 + n
    int b  = bw >> 8;
    int m  = threadIdx.x;               // 0..255
    __shared__ float lg[NW];
    const float scale = 0.08838834764831845f;   // 128^-0.5
    const float* qp = qh + (size_t)bw * CDIM;
    const float* kp = kh + ((size_t)(b * NW + m)) * CDIM;
    float acc = 0.0f;
    for (int c = 0; c < CDIM; ++c) acc += qp[c] * kp[c];
    lg[m] = acc * scale;
    __syncthreads();
    if (m == 0) {
        for (int kk = 0; kk < TOPK_; ++kk) {
            float best = -3.0e38f; int bi = 0;
            for (int j = 0; j < NW; ++j) {
                if (lg[j] > best) { best = lg[j]; bi = j; }
            }
            tidx[bw * TOPK_ + kk] = bi;
            lg[bi] = -3.0e38f;
        }
    }
}

// gather LN(xg4) tokens of the top-k global windows into sc16 rows 64..95
__global__ void k_gather_sc(const fp16_t* __restrict__ lnxg, const int* __restrict__ tidx,
                            fp16_t* __restrict__ sc16) {
    int r = blockIdx.x;                 // bw*32 + t2
    int c = threadIdx.x;
    int bw = r >> 5, t2 = r & 31;
    int b = bw >> 8;
    int kk = t2 >> 3, tt = t2 & 7;
    int widx = tidx[bw * TOPK_ + kk];
    int i1 = widx >> 6, i2 = (widx >> 3) & 7, i3 = widx & 7;
    int ms = tt >> 2, mh = (tt >> 1) & 1, mw = tt & 1;
    int tok = ((i1 * 2 + ms) * 16 + (i2 * 2 + mh)) * 16 + (i3 * 2 + mw);
    sc16[((size_t)bw * W3 + W3L + t2) * CDIM + c] =
        lnxg[((size_t)(b * GN + tok)) * CDIM + c];
}

// ---------------------------------------------------------------------------
// Global attention, flash-style online softmax. One wave per (b,h,16-row qtile).
// q/k: [B,H,2048,16] f16.  vT: [B,H,16,2048] f16.  o16: [B*2048,128] f16.
// grid = (128, 8, 2), block = 32.  K/V tiles for kt+1 prefetched; row
// reductions use DPP16 (no LDS).
// ---------------------------------------------------------------------------
__global__ void k_attn_global(const fp16_t* __restrict__ q16, const fp16_t* __restrict__ k16,
                              const fp16_t* __restrict__ vT, fp16_t* __restrict__ o16) {
    const int lane = threadIdx.x;
    const int n  = lane & 15;
    const int hs = lane >> 4;
    const int qi = blockIdx.x, h = blockIdx.y, b = blockIdx.z;
    const size_t base  = ((size_t)(b * NHEADS + h)) * GN * DH;   // q/k base
    const size_t baseV = ((size_t)(b * NHEADS + h)) * DH * GN;   // vT base
    const fp16_t* krow = k16 + base + (size_t)n * DH;            // + kt*16*DH
    const fp16_t* vrow = vT + baseV + (size_t)n * GN;            // + kt*16

    // Q fragment (K padded 16->32)
    v16h a = hs ? v16h{} : ld16(q16 + base + (size_t)(qi * 16 + n) * DH);

    v8f acc = {};
    float rmax[8], rsum[8];
#pragma unroll
    for (int r = 0; r < 8; ++r) { rmax[r] = -3.0e38f; rsum[r] = 0.0f; }

    __shared__ fp16_t ptile[16][24];    // padded rows keep 16B alignment

    const int NT = GN / 16;
    // prologue loads for kt = 0
    v16h bk = {}, bv = {};
    if (hs == 0) {
        bk = ld16(krow);
        bv = ld16(vrow);
    }

    for (int kt = 0; kt < NT; ++kt) {
        // prefetch next tile while softmax for this tile runs
        v16h bk_n = {}, bv_n = {};
        if (kt + 1 < NT && hs == 0) {
            bk_n = ld16(krow + (size_t)(kt + 1) * 16 * DH);
            bv_n = ld16(vrow + (kt + 1) * 16);
        }

        v8f z = {};
        v8f s = wmma_f16(a, bk, z);

#pragma unroll
        for (int r = 0; r < 8; ++r) {
            float sv = s[r] * 0.25f;                       // DH^-0.5
            float tm = red_max16(sv);                      // DPP row max
            float nm    = fmaxf(rmax[r], tm);
            float alpha = __expf(rmax[r] - nm);
            float pv    = __expf(sv - nm);
            float ps = red_sum16(pv);                      // DPP row sum
            rsum[r] = rsum[r] * alpha + ps;
            acc[r] *= alpha;
            rmax[r] = nm;
            ptile[r + 8 * hs][n] = (fp16_t)pv;
        }
        __syncthreads();
        // P as A fragment (keys padded 16->32): lanes 0-15 hold keys 0..15
        v16h pa = hs ? v16h{} : ld16(&ptile[n][0]);
        acc = wmma_f16(pa, bv, acc);
        __syncthreads();

        bk = bk_n; bv = bv_n;
    }

#pragma unroll
    for (int r = 0; r < 8; ++r) {
        int m = qi * 16 + r + 8 * hs;
        o16[((size_t)(b * GN + m)) * CDIM + h * DH + n] = (fp16_t)(acc[r] / rsum[r]);
    }
}

// ---------------------------------------------------------------------------
// Local (window) attention over 96 tokens; only query rows 0..63 needed.
// qkvp: [B*nw*96, 384] f16.  vloc: [bw, h, d, 96] f16.  out16: [B*nw*64,128] f16.
// grid = (512, 8, 4), block = 32
// ---------------------------------------------------------------------------
__global__ void k_attn_local(const fp16_t* __restrict__ qkvp, const fp16_t* __restrict__ vloc,
                             fp16_t* __restrict__ out16) {
    const int lane = threadIdx.x;
    const int n  = lane & 15;
    const int hs = lane >> 4;
    const int bw = blockIdx.x, h = blockIdx.y, qt = blockIdx.z;
    const float scale = 0.08838834764831845f;    // dim^-0.5 (not per-head)

    v16h a = hs ? v16h{} : ld16(qkvp + ((size_t)(bw * W3 + qt * 16 + n)) * 384 + h * DH);

    v8f sfr[6];
#pragma unroll
    for (int kt = 0; kt < 6; ++kt) {
        v16h bk = {};
        if (hs == 0)
            bk = ld16(qkvp + ((size_t)(bw * W3 + kt * 16 + n)) * 384 + CDIM + h * DH);
        v8f z = {};
        sfr[kt] = wmma_f16(a, bk, z);
    }

    // prefetch all 6 V fragments (covered by softmax below)
    const fp16_t* vbase = vloc + ((size_t)bw * CDIM + h * DH + n) * W3;
    v16h bvf[6];
#pragma unroll
    for (int kt = 0; kt < 6; ++kt) {
        v16h bv = {};
        if (hs == 0) bv = ld16(vbase + kt * 16);
        bvf[kt] = bv;
    }

    float rsum[8];
#pragma unroll
    for (int r = 0; r < 8; ++r) {
        float mx = sfr[0][r] * scale;
#pragma unroll
        for (int kt = 1; kt < 6; ++kt) mx = fmaxf(mx, sfr[kt][r] * scale);
        mx = red_max16(mx);                                // DPP row max
        float sum = 0.0f;
#pragma unroll
        for (int kt = 0; kt < 6; ++kt) {
            float p = __expf(sfr[kt][r] * scale - mx);
            sfr[kt][r] = p;
            sum += p;
        }
        rsum[r] = red_sum16(sum);                          // DPP row sum
    }

    v8f acc = {};
    __shared__ fp16_t ptile[16][24];
#pragma unroll
    for (int kt = 0; kt < 6; ++kt) {
#pragma unroll
        for (int r = 0; r < 8; ++r) ptile[r + 8 * hs][n] = (fp16_t)sfr[kt][r];
        __syncthreads();
        v16h pa = hs ? v16h{} : ld16(&ptile[n][0]);
        acc = wmma_f16(pa, bvf[kt], acc);
        __syncthreads();
    }

#pragma unroll
    for (int r = 0; r < 8; ++r) {
        int m = qt * 16 + r + 8 * hs;
        out16[((size_t)(bw * W3L + m)) * CDIM + h * DH + n] = (fp16_t)(acc[r] / rsum[r]);
    }
}

// ---------------------------------------------------------------------------
// Host launcher
// ---------------------------------------------------------------------------
extern "C" void kernel_launch(void* const* d_in, const int* in_sizes, int n_in,
                              void* d_out, int out_size, void* d_ws, size_t ws_size,
                              hipStream_t stream) {
    (void)in_sizes; (void)n_in; (void)out_size; (void)ws_size;

    const float* x_in    = (const float*)d_in[0];
    const float* x_g_in  = (const float*)d_in[1];
    const float* qkv_w   = (const float*)d_in[2];
    const float* qkv_b   = (const float*)d_in[3];
    const float* proj_w  = (const float*)d_in[4];
    const float* proj_b  = (const float*)d_in[5];
    const float* ln_g    = (const float*)d_in[6];
    const float* ln_b    = (const float*)d_in[7];
    const float* mlp_w1  = (const float*)d_in[8];
    const float* mlp_b1  = (const float*)d_in[9];
    const float* mlp_w2  = (const float*)d_in[10];
    const float* mlp_b2  = (const float*)d_in[11];
    const float* rq_w    = (const float*)d_in[12];
    const float* rq_b    = (const float*)d_in[13];
    const float* rk_w    = (const float*)d_in[14];
    const float* rk_b    = (const float*)d_in[15];
    const float* gqkv_w  = (const float*)d_in[16];
    const float* gqkv_b  = (const float*)d_in[17];
    const float* wo_w    = (const float*)d_in[18];
    const float* wo_b    = (const float*)d_in[19];
    const float* mlp2_w1 = (const float*)d_in[20];
    const float* mlp2_b1 = (const float*)d_in[21];
    const float* mlp2_w2 = (const float*)d_in[22];
    const float* mlp2_b2 = (const float*)d_in[23];

    float* out_l = (float*)d_out;                               // [2,128,16,32,32]
    float* out_g = (float*)d_out + (size_t)BATCH * CDIM * SHW;  // [2,128,8,16,16]

    // bump allocator over workspace
    char*  ws  = (char*)d_ws;
    size_t off = 0;
    auto alloc = [&](size_t bytes) -> void* {
        void* p = ws + off;
        off = (off + bytes + 255) & ~(size_t)255;
        return p;
    };

    const int NTOK_G  = BATCH * GN;           // 4096
    const int NROW_SC = BATCH * NW * W3;      // 49152
    const int NROW_L  = BATCH * NW * W3L;     // 32768

    // f16 transposed weights [N,K]
    fp16_t* qkv_wT    = (fp16_t*)alloc(sizeof(fp16_t) * 128 * 384);
    fp16_t* proj_wT   = (fp16_t*)alloc(sizeof(fp16_t) * 128 * 128);
    fp16_t* mlp_w1T   = (fp16_t*)alloc(sizeof(fp16_t) * 128 * 512);
    fp16_t* mlp_w2T   = (fp16_t*)alloc(sizeof(fp16_t) * 512 * 128);
    fp16_t* rq_wT     = (fp16_t*)alloc(sizeof(fp16_t) * 128 * 128);
    fp16_t* rk_wT     = (fp16_t*)alloc(sizeof(fp16_t) * 128 * 128);
    fp16_t* gqkv_wT   = (fp16_t*)alloc(sizeof(fp16_t) * 128 * 384);
    fp16_t* wo_wT     = (fp16_t*)alloc(sizeof(fp16_t) * 128 * 128);
    fp16_t* mlp2_w1T  = (fp16_t*)alloc(sizeof(fp16_t) * 128 * 512);
    fp16_t* mlp2_w2T  = (fp16_t*)alloc(sizeof(fp16_t) * 512 * 128);

    // activations
    float*  xg0     = (float*) alloc(sizeof(float)  * NTOK_G * CDIM);
    fp16_t* ht16    = (fp16_t*)alloc(sizeof(fp16_t) * NTOK_G * CDIM);   // reused LN outputs
    fp16_t* qkvg16  = (fp16_t*)alloc(sizeof(fp16_t) * NTOK_G * 384);
    fp16_t* q16     = (fp16_t*)alloc(sizeof(fp16_t) * NTOK_G * DH);
    fp16_t* k16     = (fp16_t*)alloc(sizeof(fp16_t) * NTOK_G * DH);
    fp16_t* vT16    = (fp16_t*)alloc(sizeof(fp16_t) * NTOK_G * DH);
    fp16_t* o16     = (fp16_t*)alloc(sizeof(fp16_t) * NTOK_G * CDIM);
    float*  xg1     = (float*) alloc(sizeof(float)  * NTOK_G * CDIM);
    fp16_t* m1_16   = (fp16_t*)alloc(sizeof(fp16_t) * NTOK_G * 512);
    float*  xg4     = (float*) alloc(sizeof(float)  * NTOK_G * CDIM);
    fp16_t* gwin16  = (fp16_t*)alloc(sizeof(fp16_t) * BATCH * NW * CDIM);
    float*  qh      = (float*) alloc(sizeof(float)  * BATCH * NW * CDIM);
    float*  kh      = (float*) alloc(sizeof(float)  * BATCH * NW * CDIM);
    int*    tidx    = (int*)   alloc(sizeof(int)    * BATCH * NW * TOPK_);
    float*  sl      = (float*) alloc(sizeof(float)  * (size_t)NROW_L * CDIM);
    fp16_t* sc16    = (fp16_t*)alloc(sizeof(fp16_t) * (size_t)NROW_SC * CDIM);
    fp16_t* qkvp16  = (fp16_t*)alloc(sizeof(fp16_t) * (size_t)NROW_SC * 384);
    fp16_t* vloc16  = (fp16_t*)alloc(sizeof(fp16_t) * (size_t)BATCH * NW * CDIM * W3);
    fp16_t* aout16  = (fp16_t*)alloc(sizeof(fp16_t) * (size_t)NROW_L * CDIM);
    float*  lbuf    = (float*) alloc(sizeof(float)  * (size_t)NROW_L * CDIM);
    fp16_t* lln16   = (fp16_t*)alloc(sizeof(fp16_t) * (size_t)NROW_L * CDIM);
    float*  lfinal  = (float*) alloc(sizeof(float)  * (size_t)NROW_L * CDIM);
    // local MLP hidden aliased onto dead qkvp16 (16.8M halfs fits in 18.9M)
    fp16_t* lm1_16  = qkvp16;

    // 1) weights -> f16 transposed
    auto cvt = [&](const float* src, fp16_t* dst, int K, int N) {
        k_cvt_t<<<dim3((K * N + 255) / 256), dim3(256), 0, stream>>>(src, dst, K, N);
    };
    cvt(qkv_w,   qkv_wT,   128, 384);
    cvt(proj_w,  proj_wT,  128, 128);
    cvt(mlp_w1,  mlp_w1T,  128, 512);
    cvt(mlp_w2,  mlp_w2T,  512, 128);
    cvt(rq_w,    rq_wT,    128, 128);
    cvt(rk_w,    rk_wT,    128, 128);
    cvt(gqkv_w,  gqkv_wT,  128, 384);
    cvt(wo_w,    wo_wT,    128, 128);
    cvt(mlp2_w1, mlp2_w1T, 128, 512);
    cvt(mlp2_w2, mlp2_w2T, 512, 128);

    // 2) channel-last token buffers
    k_build_xg0<<<dim3(NTOK_G), dim3(CDIM), 0, stream>>>(x_g_in, xg0);
    k_build_shortcut<<<dim3(NROW_L), dim3(CDIM), 0, stream>>>(x_in, sl);

    auto gemm = [&](const fp16_t* A, const fp16_t* BT, int M, int N, int K,
                    const float* bias, const float* resid,
                    float* oF, fp16_t* oH, int act) {
        dim3 g(N / 64, M / 64), blk(128);
        if (K == 128)
            k_gemm_wmma<128><<<g, blk, 0, stream>>>(A, BT, M, N, bias, resid, oF, oH, act);
        else
            k_gemm_wmma<512><<<g, blk, 0, stream>>>(A, BT, M, N, bias, resid, oF, oH, act);
    };

    // ---- global branch ----
    k_ln128<<<dim3(NTOK_G), dim3(CDIM), 0, stream>>>(xg0, ln_g, ln_b, ht16, 1, 1);
    gemm(ht16, qkv_wT, NTOK_G, 384, 128, qkv_b, nullptr, nullptr, qkvg16, 0);
    k_split_qkv<<<dim3(NTOK_G), dim3(384), 0, stream>>>(qkvg16, q16, k16, vT16);
    k_attn_global<<<dim3(GN / 16, NHEADS, BATCH), dim3(32), 0, stream>>>(q16, k16, vT16, o16);
    gemm(o16, proj_wT, NTOK_G, 128, 128, proj_b, xg0, xg1, nullptr, 0);
    k_ln128<<<dim3(NTOK_G), dim3(CDIM), 0, stream>>>(xg1, ln_g, ln_b, ht16, 1, 1);
    gemm(ht16, mlp_w1T, NTOK_G, 512, 128, mlp_b1, nullptr, nullptr, m1_16, /*gelu*/1);
    gemm(m1_16, mlp_w2T, NTOK_G, 128, 512, mlp_b2, xg1, xg4, nullptr, 0);
    k_scatter_g<<<dim3(NTOK_G), dim3(CDIM), 0, stream>>>(xg4, out_g);

    // ---- routing ----
    k_gwin_mean<<<dim3(BATCH * NW), dim3(CDIM), 0, stream>>>(xg4, gwin16);
    gemm(gwin16, rq_wT, BATCH * NW, 128, 128, rq_b, nullptr, qh, nullptr, 0);
    gemm(gwin16, rk_wT, BATCH * NW, 128, 128, rk_b, nullptr, kh, nullptr, 0);
    k_route_topk<<<dim3(BATCH * NW), dim3(NW), 0, stream>>>(qh, kh, tidx);

    // ---- build LN(sc) in f16 ----
    k_ln128<<<dim3(NTOK_G), dim3(CDIM), 0, stream>>>(xg4, ln_g, ln_b, ht16, 1, 1);   // LN(xg4)
    k_ln128<<<dim3(NROW_L), dim3(CDIM), 0, stream>>>(sl, ln_g, ln_b, sc16, W3L, W3); // local part
    k_gather_sc<<<dim3(BATCH * NW * (TOPK_ * W3G)), dim3(CDIM), 0, stream>>>(ht16, tidx, sc16);

    // ---- local+routed attention ----
    gemm(sc16, gqkv_wT, NROW_SC, 384, 128, gqkv_b, nullptr, nullptr, qkvp16, 0);
    k_repack_vloc<<<dim3(BATCH * NW), dim3(CDIM), 0, stream>>>(qkvp16, vloc16);
    k_attn_local<<<dim3(BATCH * NW, NHEADS, W3L / 16), dim3(32), 0, stream>>>(qkvp16, vloc16, aout16);
    gemm(aout16, wo_wT, NROW_L, 128, 128, wo_b, sl, lbuf, nullptr, 0);

    // ---- local FFN ----
    k_ln128<<<dim3(NROW_L), dim3(CDIM), 0, stream>>>(lbuf, ln_g, ln_b, lln16, 1, 1);
    gemm(lln16, mlp2_w1T, NROW_L, 512, 128, mlp2_b1, nullptr, nullptr, lm1_16, /*gelu*/1);
    gemm(lm1_16, mlp2_w2T, NROW_L, 128, 512, mlp2_b2, lbuf, lfinal, nullptr, 0);

    // ---- final scatter to channel-first output ----
    k_scatter_l<<<dim3(NROW_L), dim3(CDIM), 0, stream>>>(lfinal, out_l);
}
